// CausalPrefixAttention_54726473285780
// MI455X (gfx1250) — matrix operand order repaired and checked by reading, not compiled, not run on hardware
//
#include <hip/hip_runtime.h>
#include <math.h>

// ---------------------------------------------------------------------------
// CausalPrefixAttention for MI455X (gfx1250, wave32, WMMA).
// F16 operands, F32 accumulation (v_wmma_f32_16x16x32_f16).
// GEMM: B tile async-staged in LDS (global_load_async_to_lds_b128, ASYNCcnt),
// K-loop unrolled x2 with ping-pong register/buffer sets. Hand-rolled split
// barriers (s_barrier_signal/wait -1) with fused asynccnt/dscnt waits avoid
// the memory legalizer's blanket s_wait_loadcnt 0 at __syncthreads, so the
// A-register prefetch latency overlaps the barrier + LDS reads + WMMAs.
// Attention: flash-style, V/next-K fragment loads issued before the softmax
// VALU so exp/shuffle work hides global latency.
// ---------------------------------------------------------------------------

typedef __attribute__((ext_vector_type(8)))  _Float16 v8h;
typedef __attribute__((ext_vector_type(16))) _Float16 v16h;
typedef __attribute__((ext_vector_type(8)))  float    v8f;

struct H8x2 { v8h lo, hi; };

static __device__ __forceinline__ v16h joinh(v8h lo, v8h hi) {
  H8x2 t{lo, hi};
  return __builtin_bit_cast(v16h, t);
}
static __device__ __forceinline__ v8h ldg8h(const _Float16* p) {
  return *reinterpret_cast<const v8h*>(p);
}
static __device__ __forceinline__ v8f wmma_f16(v16h a, v16h b, v8f c) {
  // (neg_a, A, neg_b, B, c_mod, C, reuse_a, reuse_b)
  return __builtin_amdgcn_wmma_f32_16x16x32_f16(false, a, false, b, (short)0, c,
                                                false, false);
}
// Async 16B global -> LDS copy (per lane). ldsoff = LDS byte address
// (low 32 bits of the generic shared pointer), gsrc in VGPR pair.
static __device__ __forceinline__ void async_g2l_b128(unsigned ldsoff,
                                                      const void* gsrc) {
  asm volatile("global_load_async_to_lds_b128 %0, %1, off"
               :: "v"(ldsoff), "v"(gsrc) : "memory");
}
// Workgroup barrier: "buffer filled". Own async writes must have landed
// (<=1 outstanding: the one just issued for the other buffer) before signal.
static __device__ __forceinline__ void bar_async_filled() {
  asm volatile("s_wait_asynccnt 0x1\n\t"
               "s_barrier_signal -1\n\t"
               "s_barrier_wait -1" ::: "memory");
}
// Workgroup barrier: "reads done". Own DS reads drained before the LDS
// buffer may be overwritten by the next async fill.
static __device__ __forceinline__ void bar_reads_done() {
  asm volatile("s_wait_dscnt 0x0\n\t"
               "s_barrier_signal -1\n\t"
               "s_barrier_wait -1" ::: "memory");
}

// -------------------------- LayerNorm -> F16 --------------------------------
__global__ __launch_bounds__(256)
void ln_f16(const float* __restrict__ X, const float* __restrict__ g,
            const float* __restrict__ be, _Float16* __restrict__ out, int C) {
  const int row = blockIdx.x;
  const float* xr = X + (size_t)row * C;
  __shared__ float s1[256], s2[256];
  float a = 0.f, b2 = 0.f;
  for (int c = threadIdx.x; c < C; c += 256) { float v = xr[c]; a += v; b2 += v * v; }
  s1[threadIdx.x] = a; s2[threadIdx.x] = b2;
  __syncthreads();
  for (int ofs = 128; ofs > 0; ofs >>= 1) {
    if (threadIdx.x < ofs) {
      s1[threadIdx.x] += s1[threadIdx.x + ofs];
      s2[threadIdx.x] += s2[threadIdx.x + ofs];
    }
    __syncthreads();
  }
  const float mu = s1[0] / C;
  const float var = s2[0] / C - mu * mu;
  const float rs = rsqrtf(var + 1e-5f);
  for (int c = threadIdx.x; c < C; c += 256)
    out[(size_t)row * C + c] = (_Float16)((xr[c] - mu) * rs * g[c] + be[c]);
}

// ----------------- weight fp32 [K][N] -> f16 transposed [N][K] --------------
__global__ __launch_bounds__(256)
void wtrans(const float* __restrict__ W, _Float16* __restrict__ WT, int K, int N) {
  size_t idx = (size_t)blockIdx.x * 256 + threadIdx.x;
  if (idx >= (size_t)K * N) return;
  int k = (int)(idx / N), n = (int)(idx % N);
  WT[(size_t)n * K + k] = (_Float16)W[idx];
}

// ----------------------------- WMMA GEMM ------------------------------------
// A: [M,K] f16 row-major. BT: [N,K] f16 row-major (B transposed).
// 8 waves/block; wave computes 32x64 (2 M-tiles x 4 N-tiles, 8 accumulators);
// block = 256x64. B tile (64 cols x 32 k = 4KB) async-staged in LDS, double
// buffered; K-loop unrolled x2 (ping-pong sets, no loop-carried copies).
// mode 0: Q -> [b,h,i,d] (scaled)   mode 1: K -> [b,h,jofs+i,d]
// mode 2: VT -> [b,h,d,jofs+i]      mode 3: fp32 out [M,N] + bias
// Requires K % 64 == 0 (always 1024 here).
__global__ __launch_bounds__(256)
void gemm16(const _Float16* __restrict__ A, const _Float16* __restrict__ BT,
            _Float16* __restrict__ outH, float* __restrict__ outF,
            const float* __restrict__ bias,
            int N, int K, int mode, int jofs, float scale) {
  __shared__ _Float16 Blds[2][64 * 32];   // [buf][col*32 + k]

  const int tid  = threadIdx.x;
  const int lane = tid & 31;
  const int w    = tid >> 5;
  const int l15  = lane & 15;
  const int hi   = lane >> 4;             // 0: lanes 0-15, 1: lanes 16-31
  const int row0 = blockIdx.y * 256 + w * 32;
  const int col0 = blockIdx.x * 64;

  // Async-copy role: thread copies one 16B segment of the B tile.
  const int bcol = tid >> 2, bseg = tid & 3;
  const _Float16* bsrc = BT + (size_t)(col0 + bcol) * K + bseg * 8;
  const unsigned ldsoff0 = (unsigned)(uintptr_t)&Blds[0][bcol * 32 + bseg * 8];
  const unsigned ldsoff1 = (unsigned)(uintptr_t)&Blds[1][bcol * 32 + bseg * 8];

  const _Float16* ar0 = A + (size_t)(row0 + l15) * K + hi * 8;        // M-tile 0
  const _Float16* ar1 = A + (size_t)(row0 + 16 + l15) * K + hi * 8;   // M-tile 1

  v8f acc[2][4] = {};

  // Prologue: B(k=0) -> buf0, A(k=0) -> set A regs.
  async_g2l_b128(ldsoff0, bsrc);
  v8h a00 = ldg8h(ar0), a01 = ldg8h(ar0 + 16);
  v8h a10 = ldg8h(ar1), a11 = ldg8h(ar1 + 16);

  for (int k0 = 0; k0 < K; k0 += 64) {
    // ---- phase 0: compute k0 (setA/buf0); prefetch k0+32 (setB/buf1) -------
    const int k1 = k0 + 32;                       // < K since K % 64 == 0
    async_g2l_b128(ldsoff1, bsrc + k1);
    v8h b00 = ldg8h(ar0 + k1), b01 = ldg8h(ar0 + k1 + 16);
    v8h b10 = ldg8h(ar1 + k1), b11 = ldg8h(ar1 + k1 + 16);
    bar_async_filled();
    {
      v16h af0 = joinh(a00, a01), af1 = joinh(a10, a11);
#pragma unroll
      for (int t = 0; t < 4; ++t) {
        const _Float16* bl = &Blds[0][(t * 16 + l15) * 32 + hi * 8];
        v16h bf = joinh(*(const v8h*)bl, *(const v8h*)(bl + 16));
        acc[0][t] = wmma_f16(af0, bf, acc[0][t]);
        acc[1][t] = wmma_f16(af1, bf, acc[1][t]);
      }
    }
    bar_reads_done();
    // ---- phase 1: compute k0+32 (setB/buf1); prefetch k0+64 (setA/buf0) ----
    const int k2 = (k0 + 64 < K) ? k0 + 64 : 0;   // clamped prefetch target
    async_g2l_b128(ldsoff0, bsrc + k2);
    a00 = ldg8h(ar0 + k2); a01 = ldg8h(ar0 + k2 + 16);
    a10 = ldg8h(ar1 + k2); a11 = ldg8h(ar1 + k2 + 16);
    bar_async_filled();
    {
      v16h af0 = joinh(b00, b01), af1 = joinh(b10, b11);
#pragma unroll
      for (int t = 0; t < 4; ++t) {
        const _Float16* bl = &Blds[1][(t * 16 + l15) * 32 + hi * 8];
        v16h bf = joinh(*(const v8h*)bl, *(const v8h*)(bl + 16));
        acc[0][t] = wmma_f16(af0, bf, acc[0][t]);
        acc[1][t] = wmma_f16(af1, bf, acc[1][t]);
      }
    }
    bar_reads_done();
  }

#pragma unroll
  for (int mt = 0; mt < 2; ++mt) {
#pragma unroll
    for (int t = 0; t < 4; ++t) {
#pragma unroll
      for (int r = 0; r < 8; ++r) {
        const int orow = row0 + mt * 16 + hi * 8 + r;  // C layout: VGPR r, half-wave
        const int ocol = col0 + t * 16 + l15;
        float v = acc[mt][t][r] * scale;
        if (mode == 3) {
          outF[(size_t)orow * N + ocol] = v + bias[ocol];
        } else {
          const int b = orow >> 11, i = orow & 2047;
          const int h = ocol >> 6,  d = ocol & 63;
          if (mode == 0)
            outH[(((size_t)(b * 16 + h) * 2048) + i) * 64 + d] = (_Float16)v;
          else if (mode == 1)
            outH[(((size_t)(b * 16 + h) * 4096) + (jofs + i)) * 64 + d] = (_Float16)v;
          else
            outH[(((size_t)(b * 16 + h) * 64) + d) * 4096 + (jofs + i)] = (_Float16)v;
        }
      }
    }
  }
}

// --------------------------- Flash attention --------------------------------
// Q: [b,h,2048,64], K: [b,h,4096,64], VT: [b,h,64,4096], all f16.
// Wave owns 16 queries x full d=64 accumulator; key blocks of 32.
// Causal rule: key jj visible to query ii iff jj <= ii + 2048.
__global__ __launch_bounds__(256)
void attn(const _Float16* __restrict__ Q, const _Float16* __restrict__ Kt,
          const _Float16* __restrict__ VT, const unsigned char* __restrict__ cmask,
          _Float16* __restrict__ O) {
  const int lane = threadIdx.x & 31;
  const int w    = threadIdx.x >> 5;
  const int l15  = lane & 15;
  const int hi   = lane >> 4;
  const int b = blockIdx.z, h = blockIdx.y;
  const int q0 = blockIdx.x * 128 + w * 16;

  const _Float16* Qbh = Q  + (size_t)(b * 16 + h) * 2048 * 64;
  const _Float16* Kbh = Kt + (size_t)(b * 16 + h) * 4096 * 64;
  const _Float16* Vbh = VT + (size_t)(b * 16 + h) * 64 * 4096;
  const unsigned char* cm = cmask + (size_t)b * 2048;

  __shared__ _Float16 Plds[8][16][32];   // per-wave P staging (1KB each)

  // Q fragments (16 rows x d=64 -> two K=32 A-frags), live across the loop.
  v16h qf[2];
#pragma unroll
  for (int t = 0; t < 2; ++t) {
    const _Float16* ar = Qbh + (size_t)(q0 + l15) * 64 + t * 32 + hi * 8;
    qf[t] = joinh(ldg8h(ar), ldg8h(ar + 16));
  }

  // kf[t*2+sb]: B-frag of K for d-chunk t, key subtile sb.
  auto loadK = [&](int j0, v16h* dst) {
#pragma unroll
    for (int t = 0; t < 2; ++t)
#pragma unroll
      for (int sb = 0; sb < 2; ++sb) {
        const _Float16* br =
            Kbh + (size_t)(j0 + sb * 16 + l15) * 64 + t * 32 + hi * 8;
        dst[t * 2 + sb] = joinh(ldg8h(br), ldg8h(br + 16));
      }
  };

  v8f o[4] = {};
  float mrow[8], lrow[8];
#pragma unroll
  for (int r = 0; r < 8; ++r) { mrow[r] = -3.0e38f; lrow[r] = 0.f; }

  int jend = q0 + 15 + 2048 + 1;          // exclusive bound over keys
  jend = (jend + 31) & ~31;
  if (jend > 4096) jend = 4096;

  v16h kf[4];
  loadK(0, kf);

  for (int j0 = 0; j0 < jend; j0 += 32) {
    // ---- S = Q * K^T (fragments already resident) --------------------------
    v8f s0 = {}, s1 = {};
    s0 = wmma_f16(qf[0], kf[0], s0);
    s1 = wmma_f16(qf[0], kf[1], s1);
    s0 = wmma_f16(qf[1], kf[2], s0);
    s1 = wmma_f16(qf[1], kf[3], s1);

    // ---- issue V frags + next K frags BEFORE softmax (latency hiding) ------
    v16h vf[4];
#pragma unroll
    for (int t = 0; t < 4; ++t) {
      const _Float16* vr = Vbh + (size_t)(t * 16 + l15) * 4096 + j0 + hi * 8;
      vf[t] = joinh(ldg8h(vr), ldg8h(vr + 16));
    }
    v16h kfn[4];
    const int jn = (j0 + 32 < jend) ? j0 + 32 : j0;
    loadK(jn, kfn);

    // ---- masking + online softmax (hides the loads above) ------------------
    const int  jj0 = j0 + l15, jj1 = j0 + 16 + l15;
    const bool kv0 = (jj0 >= 2048) ? true : (cm[jj0] != 0);
    const bool kv1 = (jj1 >= 2048) ? true : (cm[jj1] != 0);
#pragma unroll
    for (int r = 0; r < 8; ++r) {
      const int ii = q0 + hi * 8 + r;
      float v0 = s0[r], v1 = s1[r];
      if (!kv0 || jj0 > ii + 2048) v0 = -1.0e30f;
      if (!kv1 || jj1 > ii + 2048) v1 = -1.0e30f;
      float mx = fmaxf(v0, v1);
#pragma unroll
      for (int msk = 8; msk; msk >>= 1) mx = fmaxf(mx, __shfl_xor(mx, msk, 32));
      const float mnew  = fmaxf(mrow[r], mx);
      const float alpha = __expf(mrow[r] - mnew);
      const float p0 = __expf(v0 - mnew);
      const float p1 = __expf(v1 - mnew);
      float ps = p0 + p1;
#pragma unroll
      for (int msk = 8; msk; msk >>= 1) ps += __shfl_xor(ps, msk, 32);
      lrow[r] = lrow[r] * alpha + ps;
      mrow[r] = mnew;
#pragma unroll
      for (int t = 0; t < 4; ++t) o[t][r] *= alpha;
      Plds[w][hi * 8 + r][l15]      = (_Float16)p0;
      Plds[w][hi * 8 + r][16 + l15] = (_Float16)p1;
    }
    // Cross-lane LDS dependency within the wave: order + drain DS.
    asm volatile("s_wait_dscnt 0" ::: "memory");

    // ---- P as A-frag, O += P * V (V frags already resident) ----------------
    const _Float16* pr = &Plds[w][l15][hi * 8];
    v16h pf = joinh(*(const v8h*)pr, *(const v8h*)(pr + 16));
#pragma unroll
    for (int t = 0; t < 4; ++t) o[t] = wmma_f16(pf, vf[t], o[t]);

#pragma unroll
    for (int t = 0; t < 4; ++t) kf[t] = kfn[t];
  }

  // ---- normalize + write [b, i, h*64+d] f16 --------------------------------
#pragma unroll
  for (int r = 0; r < 8; ++r) {
    const int ii = q0 + hi * 8 + r;
    const float inv = 1.f / fmaxf(lrow[r], 1e-30f);
#pragma unroll
    for (int t = 0; t < 4; ++t)
      O[((size_t)b * 2048 + ii) * 1024 + h * 64 + t * 16 + l15] =
          (_Float16)(o[t][r] * inv);
  }
}

// ----------------------------- host launcher --------------------------------
extern "C" void kernel_launch(void* const* d_in, const int* in_sizes, int n_in,
                              void* d_out, int out_size, void* d_ws, size_t ws_size,
                              hipStream_t stream) {
  const float* x    = (const float*)d_in[0];
  const float* ctx  = (const float*)d_in[1];
  const unsigned char* cmask = (const unsigned char*)d_in[2];
  const float* g1 = (const float*)d_in[3];
  const float* b1 = (const float*)d_in[4];
  const float* g2 = (const float*)d_in[5];
  const float* b2 = (const float*)d_in[6];
  const float* Wq  = (const float*)d_in[7];
  const float* Wkv = (const float*)d_in[8];
  const float* Wo  = (const float*)d_in[9];
  const float* bo  = (const float*)d_in[10];
  float* out = (float*)d_out;

  char* p = (char*)d_ws;
  _Float16* XN   = (_Float16*)p; p += (size_t)4096 * 1024 * 2;
  _Float16* CN   = (_Float16*)p; p += (size_t)4096 * 1024 * 2;
  _Float16* WqT  = (_Float16*)p; p += (size_t)1024 * 1024 * 2;
  _Float16* WkvT = (_Float16*)p; p += (size_t)2048 * 1024 * 2;
  _Float16* WoT  = (_Float16*)p; p += (size_t)1024 * 1024 * 2;
  _Float16* Qb   = (_Float16*)p; p += (size_t)2 * 16 * 2048 * 64 * 2;
  _Float16* Kb   = (_Float16*)p; p += (size_t)2 * 16 * 4096 * 64 * 2;
  _Float16* VTb  = (_Float16*)p; p += (size_t)2 * 16 * 64 * 4096 * 2;
  _Float16* Ob   = (_Float16*)p; p += (size_t)4096 * 1024 * 2;

  ln_f16<<<4096, 256, 0, stream>>>(x,   g1, b1, XN, 1024);
  ln_f16<<<4096, 256, 0, stream>>>(ctx, g2, b2, CN, 1024);
  wtrans<<<(1024 * 1024 + 255) / 256, 256, 0, stream>>>(Wq,  WqT,  1024, 1024);
  wtrans<<<(1024 * 2048 + 255) / 256, 256, 0, stream>>>(Wkv, WkvT, 1024, 2048);
  wtrans<<<(1024 * 1024 + 255) / 256, 256, 0, stream>>>(Wo,  WoT,  1024, 1024);

  dim3 gg(16, 16);  // N/64 x M/256
  const float qscale = 0.125f;  // 64^-0.5
  gemm16<<<gg, 256, 0, stream>>>(XN, WqT, Qb, nullptr, nullptr, 1024, 1024, 0, 0, qscale);
  gemm16<<<gg, 256, 0, stream>>>(XN, WkvT, Kb, nullptr, nullptr, 1024, 1024, 1, 2048, 1.f);
  gemm16<<<gg, 256, 0, stream>>>(CN, WkvT, Kb, nullptr, nullptr, 1024, 1024, 1, 0, 1.f);
  gemm16<<<gg, 256, 0, stream>>>(XN, WkvT + (size_t)1024 * 1024, VTb, nullptr, nullptr,
                                 1024, 1024, 2, 2048, 1.f);
  gemm16<<<gg, 256, 0, stream>>>(CN, WkvT + (size_t)1024 * 1024, VTb, nullptr, nullptr,
                                 1024, 1024, 2, 0, 1.f);

  attn<<<dim3(16, 16, 2), 256, 0, stream>>>(Qb, Kb, VTb, cmask, Ob);

  gemm16<<<gg, 256, 0, stream>>>(Ob, WoT, nullptr, out, bo, 1024, 1024, 3, 0, 1.f);
}
